// GraphEncoder_35811437314143
// MI455X (gfx1250) — matrix operand order repaired and checked by reading, not compiled, not run on hardware
//
#include <hip/hip_runtime.h>
#include <hip/hip_bf16.h>

#define DH 128
#define DLAT 64
#define NGRAPHS 2048
#define NLAYERS 3

typedef __attribute__((ext_vector_type(16))) __bf16 v16bf;
typedef __attribute__((ext_vector_type(8)))  float  v8f;

// ---------------------------------------------------------------------------
// WMMA helper: D = A(16x32 bf16) * B(32x16 bf16) + C(16x16 f32)
// ---------------------------------------------------------------------------
__device__ __forceinline__ v8f wmma_bf16(v16bf a, v16bf b, v8f c) {
    return __builtin_amdgcn_wmma_f32_16x16x32_bf16(
        /*neg_a=*/false, a, /*neg_b=*/false, b,
        /*c_mod=*/(short)0, c, /*reuse_a=*/false, /*reuse_b=*/false);
}

// A-fragment from a global f32 row. 16-bit A 16x32 layout used by this code (contiguous k):
// lane L: row m = L%16, kb = kstep*32 + (L<16 ? 0 : 16)
//   elements 0..15 -> K = kb + j (contiguous, matches B)
__device__ __forceinline__ void load_a_global(const float* __restrict__ arow,
                                              int kb, v16bf& a) {
    const float4 p0 = *(const float4*)(arow + kb);
    const float4 p1 = *(const float4*)(arow + kb + 4);
    const float4 p2 = *(const float4*)(arow + kb + 8);
    const float4 p3 = *(const float4*)(arow + kb + 12);
    a[0]=(__bf16)p0.x;  a[1]=(__bf16)p0.y;  a[2]=(__bf16)p0.z;  a[3]=(__bf16)p0.w;
    a[4]=(__bf16)p1.x;  a[5]=(__bf16)p1.y;  a[6]=(__bf16)p1.z;  a[7]=(__bf16)p1.w;
    a[8]=(__bf16)p2.x;  a[9]=(__bf16)p2.y;  a[10]=(__bf16)p2.z; a[11]=(__bf16)p2.w;
    a[12]=(__bf16)p3.x; a[13]=(__bf16)p3.y; a[14]=(__bf16)p3.z; a[15]=(__bf16)p3.w;
}

// A-fragment from an LDS bf16 row (same K mapping as above)
__device__ __forceinline__ void load_a_lds(const __bf16* trow, int kb, v16bf& a) {
#pragma unroll
    for (int j = 0; j < 8; ++j) a[j] = trow[kb + j];
#pragma unroll
    for (int j = 0; j < 8; ++j) a[8 + j] = trow[kb + 8 + j];
}

// B-fragment from LDS W^T[n][k] (bf16). 16-bit B 32x16 layout used by this code (contiguous k):
// lane L: col n = L%16, K = kstep*32 + (L<16 ? 0 : 16) + j, j = 0..15 contiguous.
__device__ __forceinline__ v16bf load_b_lds(const __bf16* __restrict__ p) {
    v16bf b;
#pragma unroll
    for (int j = 0; j < 16; ++j) b[j] = p[j];
    return b;
}

// ---------------------------------------------------------------------------
// Elementwise / scatter / reduce kernels
// ---------------------------------------------------------------------------
__global__ void zero_f32(float* __restrict__ p, long n) {
    long i = (long)blockIdx.x * blockDim.x + threadIdx.x;
    if (i < n) p[i] = 0.0f;
}

__global__ void copy_f32v4(const float* __restrict__ in, float* __restrict__ out, long n4) {
    long i = (long)blockIdx.x * blockDim.x + threadIdx.x;
    if (i < n4) ((float4*)out)[i] = ((const float4*)in)[i];
}

// z[dst[e]] += h[src[e]]  (one edge per 32 lanes, 4 features per lane)
__global__ void scatter_add_edges(const float* __restrict__ h,
                                  const int* __restrict__ src,
                                  const int* __restrict__ dst,
                                  float* __restrict__ z, int n_edges) {
    int t = blockIdx.x * blockDim.x + threadIdx.x;
    int e = t >> 5, lane = t & 31;
    if (e >= n_edges) return;
    int s = src[e], d = dst[e];
    float4 v = ((const float4*)(h + (long)s * DH))[lane];
    float* p = z + (long)d * DH + lane * 4;
    atomicAdd(p + 0, v.x);
    atomicAdd(p + 1, v.y);
    atomicAdd(p + 2, v.z);
    atomicAdd(p + 3, v.w);
}

// per-feature sum and sum-of-squares into stats[0..127] / stats[128..255]
__global__ void bn_stats(const float* __restrict__ Z, float* __restrict__ stats, int n_rows) {
    int f = threadIdx.x & 127;
    int sub = threadIdx.x >> 7;           // 0 or 1
    float s = 0.0f, s2 = 0.0f;
    for (long r = (long)blockIdx.x * 2 + sub; r < n_rows; r += (long)gridDim.x * 2) {
        float v = Z[r * DH + f];
        s += v; s2 += v * v;
    }
    __shared__ float red[512];
    red[threadIdx.x] = s;
    red[256 + threadIdx.x] = s2;
    __syncthreads();
    if (sub == 0) {
        atomicAdd(&stats[f],      red[f]       + red[128 + f]);
        atomicAdd(&stats[DH + f], red[256 + f] + red[384 + f]);
    }
}

// h = relu((z - mean) * rsqrt(var+eps) * gamma + beta)
__global__ void bn_apply(const float* __restrict__ Z, float* __restrict__ H,
                         const float* __restrict__ stats,
                         const float* __restrict__ gamma,
                         const float* __restrict__ beta, int n_rows) {
    long i = (long)blockIdx.x * blockDim.x + threadIdx.x;
    long total = (long)n_rows * DH;
    if (i >= total) return;
    int f = (int)(i & 127);
    float inv_n = 1.0f / (float)n_rows;
    float mean = stats[f] * inv_n;
    float var  = stats[DH + f] * inv_n - mean * mean;
    float v = (Z[i] - mean) * rsqrtf(var + 1e-5f) * gamma[f] + beta[f];
    H[i] = v > 0.0f ? v : 0.0f;
}

// segment-sum into sums[g] + counts[g]
__global__ void pool_accum(const float* __restrict__ H, const int* __restrict__ batch,
                           float* __restrict__ sums, float* __restrict__ counts, int n_rows) {
    int t = blockIdx.x * blockDim.x + threadIdx.x;
    int node = t >> 5, lane = t & 31;
    if (node >= n_rows) return;
    int g = batch[node];
    float4 v = ((const float4*)(H + (long)node * DH))[lane];
    float* p = sums + (long)g * DH + lane * 4;
    atomicAdd(p + 0, v.x);
    atomicAdd(p + 1, v.y);
    atomicAdd(p + 2, v.z);
    atomicAdd(p + 3, v.w);
    if (lane == 0) atomicAdd(&counts[g], 1.0f);
}

__global__ void pool_final(float* __restrict__ pooled, const float* __restrict__ counts,
                           int n_graphs) {
    int i = blockIdx.x * blockDim.x + threadIdx.x;
    if (i >= n_graphs * DH) return;
    int g = i >> 7;
    float c = counts[g];
    c = c > 1.0f ? c : 1.0f;
    pooled[i] = pooled[i] / c;
}

// ---------------------------------------------------------------------------
// Fused GIN MLP: Zout = relu(Z @ W1 + b1) @ W2 + b2   (in place, Zout == Z ok)
// 256 threads = 8 waves; 128 rows per block; hidden activation kept in LDS bf16.
// ---------------------------------------------------------------------------
__launch_bounds__(256)
__global__ void gin_mlp(const float* __restrict__ Z, float* __restrict__ Zout,
                        const float* __restrict__ W1, const float* __restrict__ b1,
                        const float* __restrict__ W2, const float* __restrict__ b2,
                        int n_rows) {
    __shared__ __bf16 sW1t[DH * DH];      // W1^T [n][k]
    __shared__ __bf16 sW2t[DH * DH];      // W2^T [n][k]
    __shared__ float  sB1[DH], sB2[DH];
    __shared__ __bf16 sT[8 * 16 * DH];    // per-wave 16x128 hidden tile

    int tid = threadIdx.x;
    for (int i = tid; i < DH * DH; i += 256) {
        int k = i >> 7, n = i & 127;
        sW1t[n * DH + k] = (__bf16)W1[i];
        sW2t[n * DH + k] = (__bf16)W2[i];
    }
    if (tid < DH) { sB1[tid] = b1[tid]; sB2[tid] = b2[tid]; }
    __syncthreads();

    int wave = tid >> 5;
    int lane = tid & 31;
    int m    = lane & 15;
    int hi   = lane >> 4;                 // lane group (0: lanes 0-15, 1: 16-31)
    int row0 = blockIdx.x * 128 + wave * 16;

    long arow_i = (long)row0 + m;
    if (arow_i >= n_rows) arow_i = n_rows - 1;      // clamp loads; stores guarded
    const float* arow = Z + arow_i * DH;

    // ---- phase 1: A from global f32, B = W1^T, hidden -> LDS bf16 ----
    v16bf a[4];
#pragma unroll
    for (int kt = 0; kt < 4; ++kt)
        load_a_global(arow, kt * 32 + hi * 16, a[kt]);

    __bf16* myT = sT + wave * 16 * DH;
#pragma unroll
    for (int nt = 0; nt < 8; ++nt) {
        v8f c = {};
#pragma unroll
        for (int kt = 0; kt < 4; ++kt) {
            v16bf b = load_b_lds(sW1t + (nt * 16 + m) * DH + kt * 32 + hi * 16);
            c = wmma_bf16(a[kt], b, c);
        }
        float bias = sB1[nt * 16 + m];
#pragma unroll
        for (int r = 0; r < 8; ++r) {
            float v = c[r] + bias;
            v = v > 0.0f ? v : 0.0f;
            myT[(r + hi * 8) * DH + nt * 16 + m] = (__bf16)v;   // row r+8*hi, col nt*16+m
        }
    }

    // ---- phase 2: A from LDS hidden tile, B = W2^T, store f32 ----
#pragma unroll
    for (int kt = 0; kt < 4; ++kt)
        load_a_lds(myT + m * DH, kt * 32 + hi * 16, a[kt]);

#pragma unroll
    for (int nt = 0; nt < 8; ++nt) {
        v8f c = {};
#pragma unroll
        for (int kt = 0; kt < 4; ++kt) {
            v16bf b = load_b_lds(sW2t + (nt * 16 + m) * DH + kt * 32 + hi * 16);
            c = wmma_bf16(a[kt], b, c);
        }
        float bias = sB2[nt * 16 + m];
#pragma unroll
        for (int r = 0; r < 8; ++r) {
            long rr = (long)row0 + r + hi * 8;
            if (rr < n_rows)
                Zout[rr * DH + nt * 16 + m] = c[r] + bias;
        }
    }
}

// ---------------------------------------------------------------------------
// Head GEMM: out[M x n_cols] = A[M x 128] @ W[128 x n_cols] + bias  (n_cols<=64)
// ---------------------------------------------------------------------------
__launch_bounds__(256)
__global__ void gemm_head(const float* __restrict__ A, const float* __restrict__ W,
                          const float* __restrict__ bias, float* __restrict__ out,
                          int n_rows, int n_cols) {
    __shared__ __bf16 sWt[DLAT * DH];     // W^T [n][k]
    __shared__ float  sB[DLAT];

    int tid = threadIdx.x;
    for (int i = tid; i < DH * n_cols; i += 256) {
        int k = i / n_cols, n = i % n_cols;
        sWt[n * DH + k] = (__bf16)W[i];
    }
    if (tid < n_cols) sB[tid] = bias[tid];
    __syncthreads();

    int wave = tid >> 5;
    int lane = tid & 31;
    int m    = lane & 15;
    int hi   = lane >> 4;
    int row0 = blockIdx.x * 128 + wave * 16;

    long arow_i = (long)row0 + m;
    if (arow_i >= n_rows) arow_i = n_rows - 1;
    const float* arow = A + arow_i * DH;

    v16bf a[4];
#pragma unroll
    for (int kt = 0; kt < 4; ++kt)
        load_a_global(arow, kt * 32 + hi * 16, a[kt]);

    int ntiles = n_cols >> 4;
    for (int nt = 0; nt < ntiles; ++nt) {
        v8f c = {};
#pragma unroll
        for (int kt = 0; kt < 4; ++kt) {
            v16bf b = load_b_lds(sWt + (nt * 16 + m) * DH + kt * 32 + hi * 16);
            c = wmma_bf16(a[kt], b, c);
        }
        float bv = sB[nt * 16 + m];
#pragma unroll
        for (int r = 0; r < 8; ++r) {
            long rr = (long)row0 + r + hi * 8;
            if (rr < n_rows)
                out[rr * n_cols + nt * 16 + m] = c[r] + bv;
        }
    }
}

// ---------------------------------------------------------------------------
// Host launch
// ---------------------------------------------------------------------------
extern "C" void kernel_launch(void* const* d_in, const int* in_sizes, int n_in,
                              void* d_out, int out_size, void* d_ws, size_t ws_size,
                              hipStream_t stream) {
    const float* x     = (const float*)d_in[0];
    const int*   ei    = (const int*)d_in[1];
    const int*   batch = (const int*)d_in[2];
    const float* W1    = (const float*)d_in[3];
    const float* b1    = (const float*)d_in[4];
    const float* W2    = (const float*)d_in[5];
    const float* b2    = (const float*)d_in[6];
    const float* gamma = (const float*)d_in[7];
    const float* beta  = (const float*)d_in[8];
    const float* Wmu   = (const float*)d_in[9];
    const float* bmu   = (const float*)d_in[10];
    const float* Wlv   = (const float*)d_in[11];
    const float* blv   = (const float*)d_in[12];
    float* out = (float*)d_out;

    const int N = in_sizes[0] / DH;       // 100000 nodes
    const int E = in_sizes[1] / 2;        // 400000 edges
    const int G = NGRAPHS;
    const int* src = ei;
    const int* dst = ei + E;

    // workspace layout
    float* hbuf   = (float*)d_ws;                     // N*DH
    float* zbuf   = hbuf + (size_t)N * DH;            // N*DH
    float* pooled = zbuf + (size_t)N * DH;            // G*DH
    float* counts = pooled + (size_t)G * DH;          // G
    float* stats  = counts + G;                       // 2*DH

    const long nelem  = (long)N * DH;
    const int  blkCpy = (int)((nelem / 4 + 255) / 256);
    const int  blkEw  = (int)((nelem + 255) / 256);
    const int  blkEdg = (int)(((long)E * 32 + 255) / 256);
    const int  blkMlp = (N + 127) / 128;
    const int  blkPl  = (int)(((long)N * 32 + 255) / 256);

    const float* h = x;
    for (int l = 0; l < NLAYERS; ++l) {
        // z = h ; z[dst] += h[src]
        copy_f32v4<<<blkCpy, 256, 0, stream>>>(h, zbuf, nelem / 4);
        scatter_add_edges<<<blkEdg, 256, 0, stream>>>(h, src, dst, zbuf, E);
        // z = relu(z@W1+b1)@W2+b2   (in place, WMMA)
        gin_mlp<<<blkMlp, 256, 0, stream>>>(zbuf, zbuf,
                                            W1 + (size_t)l * DH * DH, b1 + (size_t)l * DH,
                                            W2 + (size_t)l * DH * DH, b2 + (size_t)l * DH, N);
        // batchnorm (training stats) + relu -> hbuf
        zero_f32<<<1, 256, 0, stream>>>(stats, 2 * DH);
        bn_stats<<<512, 256, 0, stream>>>(zbuf, stats, N);
        bn_apply<<<blkEw, 256, 0, stream>>>(zbuf, hbuf, stats,
                                            gamma + (size_t)l * DH, beta + (size_t)l * DH, N);
        h = hbuf;
    }

    // global mean pool
    zero_f32<<<(G * DH + G + 255) / 256, 256, 0, stream>>>(pooled, (long)G * DH + G);
    pool_accum<<<blkPl, 256, 0, stream>>>(hbuf, batch, pooled, counts, N);
    pool_final<<<(G * DH + 255) / 256, 256, 0, stream>>>(pooled, counts, G);

    // heads: mu / logvar (WMMA)
    const int blkHead = (G + 127) / 128;
    gemm_head<<<blkHead, 256, 0, stream>>>(pooled, Wmu, bmu, out, G, DLAT);
    gemm_head<<<blkHead, 256, 0, stream>>>(pooled, Wlv, blv, out + (size_t)G * DLAT, G, DLAT);
}
